// Encoder_82059645157846
// MI455X (gfx1250) — compile-verified, matrix operand order checked by reading, not run on hardware
//
#include <hip/hip_runtime.h>

typedef __attribute__((ext_vector_type(16))) __bf16        v16bf;
typedef __attribute__((ext_vector_type(8)))  float         v8f;
typedef __attribute__((ext_vector_type(4)))  float         f4;
typedef __attribute__((ext_vector_type(4)))  unsigned int  u4;

union BFrag { u4 u[2]; v16bf v; };

struct FragSet { BFrag b; BFrag a[4]; };   // one k-step of operands (40 VGPRs)

#define S_LEN 512
#define BATCH 64
#define EMB   256
#define HID   512
#define KTOT  768   // EMB + HID
#define G4    2048  // 4*HID
#define NKB   24    // KTOT / 32

// ---- helpers -------------------------------------------------------------

__device__ __forceinline__ unsigned short f2bf(float f) {
  unsigned int x = __float_as_uint(f);
  x = x + 0x7FFFu + ((x >> 16) & 1u);   // round-to-nearest-even
  return (unsigned short)(x >> 16);
}

__device__ __forceinline__ u4 pack8(const float* p) {
  f4 a = *(const f4*)p;
  f4 b = *(const f4*)(p + 4);
  u4 r;
  r.x = (unsigned)f2bf(a.x) | ((unsigned)f2bf(a.y) << 16);
  r.y = (unsigned)f2bf(a.z) | ((unsigned)f2bf(a.w) << 16);
  r.z = (unsigned)f2bf(b.x) | ((unsigned)f2bf(b.y) << 16);
  r.w = (unsigned)f2bf(b.z) | ((unsigned)f2bf(b.w) << 16);
  return r;
}

__device__ __forceinline__ float fsigmoid(float x) {
  return 1.0f / (1.0f + __expf(-x));
}
__device__ __forceinline__ float ftanh(float x) {
  return 1.0f - 2.0f / (__expf(2.0f * x) + 1.0f);
}

// load one k-step's fragments; kb is a compile-time constant under full unroll
__device__ __forceinline__ void load_set(FragSet& s,
    const unsigned short* __restrict__ Browl,  // W row base  (+ lh*8 folded in)
    const unsigned short* __restrict__ Ael,    // embS + lm*EMB + lh*8
    const unsigned short* __restrict__ Ahl,    // hin  + lm*HID + lh*8
    int kb) {
  s.b.u[0] = *(const u4*)(Browl + kb);
  s.b.u[1] = *(const u4*)(Browl + kb + 16);
  if (kb < EMB) {
    #pragma unroll
    for (int rt = 0; rt < 4; ++rt) {
      const unsigned short* ap = Ael + rt * 16 * EMB + kb;
      s.a[rt].u[0] = *(const u4*)ap;
      s.a[rt].u[1] = *(const u4*)(ap + 16);
    }
  } else {
    #pragma unroll
    for (int rt = 0; rt < 4; ++rt) {
      const unsigned short* ap = Ahl + rt * 16 * HID + (kb - EMB);
      s.a[rt].u[0] = *(const u4*)ap;
      s.a[rt].u[1] = *(const u4*)(ap + 16);
    }
  }
}

// ---- prep: W = [W_ih | W_hh] -> bf16 [2048][768]; bias = b_ih + b_hh ------

__global__ __launch_bounds__(256) void prep_w(
    const float* __restrict__ Wih, const float* __restrict__ Whh,
    const float* __restrict__ bih, const float* __restrict__ bhh,
    unsigned short* __restrict__ Wbf, float* __restrict__ bias) {
  int idx = blockIdx.x * 256 + threadIdx.x;      // < 2048*96
  int g = idx / 96;
  int k = (idx % 96) * 8;
  const float* src = (k < EMB) ? (Wih + (size_t)g * EMB + k)
                               : (Whh + (size_t)g * HID + (k - EMB));
  *(u4*)(Wbf + (size_t)g * KTOT + k) = pack8(src);
  if (idx < G4) bias[idx] = bih[idx] + bhh[idx];
}

// ---- prep: gather embeddings for all (s,b), convert to bf16 ---------------

__global__ __launch_bounds__(256) void prep_emb(
    const int* __restrict__ seq, const float* __restrict__ emb,
    unsigned short* __restrict__ ebf) {
  int idx = blockIdx.x * 256 + threadIdx.x;      // < 512*64*32
  int s   = idx >> 11;
  int rem = idx & 2047;
  int b   = rem >> 5;
  int e8  = (rem & 31) << 3;
  int tok = seq[b * S_LEN + s];
  const float* src = emb + (size_t)tok * EMB + e8;
  *(u4*)(ebf + ((size_t)(s * BATCH + b)) * EMB + e8) = pack8(src);
}

// ---- init h (both buffers) and c to zero ----------------------------------

__global__ __launch_bounds__(256) void init_state(
    unsigned int* __restrict__ h0, unsigned int* __restrict__ h1,
    float* __restrict__ c) {
  int idx = blockIdx.x * 256 + threadIdx.x;      // 32768
  c[idx] = 0.0f;
  if (idx < 16384) { h0[idx] = 0u; h1[idx] = 0u; }
}

// ---- one LSTM timestep: gates = [emb_s|h] @ Wbf^T + bias; state update ----
// grid = 16 WGs (32 hidden cols each), block = 256 (8 wave32).
// Barrier-free K loop; B/W fragments pipelined by the scheduler, A-panel rows
// prefetched into WGP$ so the just-in-time A loads are near-hits.

__global__ __launch_bounds__(256) void lstm_step(
    const unsigned short* __restrict__ embS,   // [64][256] bf16 for step s
    const unsigned short* __restrict__ hin,    // [64][512] bf16
    const unsigned short* __restrict__ Wbf,    // [2048][768] bf16
    const float*          __restrict__ bias,   // [2048]
    float*                __restrict__ cst,    // [64][512] f32 state
    unsigned short*       __restrict__ hout_bf,// [64][512] bf16 (next h)
    float*                __restrict__ h_out,  // d_out h region
    float*                __restrict__ c_out)  // d_out c region
{
  __shared__ __align__(16) float Gh[32 * 136]; // gate combine scratch

  const int tid  = threadIdx.x;
  const int w    = tid >> 5;        // wave 0..7
  const int lane = tid & 31;
  const int lm   = lane & 15;
  const int lh   = lane >> 4;
  const int hb   = blockIdx.x << 5; // hidden base 0..480

  // this wave's 16 output columns inside the 128-wide gate block
  const int lc = (w << 4) | lm;     // 0..127 : [0,32)=i [32,64)=f [64,96)=g [96,128)=o
  const int q  = lc >> 5;
  const int cc = lc & 31;
  const unsigned short* Brow  = Wbf + (size_t)(q * HID + hb + cc) * KTOT;
  const unsigned short* Browl = Brow + lh * 8;

  // warm WGP$ with this lane's W row (1536 B = 6 x 256 B lines)
  #pragma unroll
  for (int p = 0; p < 6; ++p) __builtin_prefetch(Brow + p * 128, 0, 1);

  // wave 0: warm WGP$ with the whole A panel (all 64 rows of emb_s and h);
  // 16 lanes x 4 row-tiles cover every row, prefetch has no counter cost
  if (w == 0) {
    #pragma unroll
    for (int rt = 0; rt < 4; ++rt) {
      const unsigned short* er = embS + (rt * 16 + lm) * EMB;  // 512 B row
      __builtin_prefetch(er, 0, 1);
      __builtin_prefetch(er + 128, 0, 1);
      const unsigned short* hr = hin + (rt * 16 + lm) * HID;   // 1024 B row
      __builtin_prefetch(hr, 0, 1);
      __builtin_prefetch(hr + 128, 0, 1);
      __builtin_prefetch(hr + 256, 0, 1);
      __builtin_prefetch(hr + 384, 0, 1);
    }
  }

  // per-lane A fragment bases (row = rt*16 + lm, k-chunks at lh*8 / 16+lh*8)
  const unsigned short* Ael = embS + lm * EMB + lh * 8;
  const unsigned short* Ahl = hin  + lm * HID + lh * 8;

  v8f acc[4] = {};                  // 4 row tiles x (16x16 f32)

  FragSet fs[2];
  load_set(fs[0], Browl, Ael, Ahl, 0);            // prologue: k-step 0

  #pragma unroll
  for (int it = 0; it < NKB; ++it) {
    if (it + 1 < NKB)                              // issue next set's loads
      load_set(fs[(it + 1) & 1], Browl, Ael, Ahl, (it + 1) * 32);
    const FragSet& c = fs[it & 1];                 // consume current set
    #pragma unroll
    for (int rt = 0; rt < 4; ++rt)
      acc[rt] = __builtin_amdgcn_wmma_f32_16x16x32_bf16(
          false, c.a[rt].v, false, c.b.v, (short)0, acc[rt], false, false);
  }

  // combine gates across waves via LDS, 32 batch rows at a time
  #pragma unroll
  for (int hh = 0; hh < 2; ++hh) {
    __syncthreads();
    #pragma unroll
    for (int rt2 = 0; rt2 < 2; ++rt2) {
      const v8f a = acc[hh * 2 + rt2];
      #pragma unroll
      for (int v = 0; v < 8; ++v) {
        int row = rt2 * 16 + lh * 8 + v;           // C/D layout: M = v + 8*(lane/16)
        Gh[row * 136 + lc] = a[v];
      }
    }
    __syncthreads();

    #pragma unroll
    for (int it = 0; it < 4; ++it) {
      int idx  = it * 256 + tid;                   // 0..1023
      int lr   = idx >> 5;                         // local batch row 0..31
      int c2   = idx & 31;                         // hidden col offset 0..31
      int brow = hh * 32 + lr;
      int hcol = hb + c2;
      float gi = Gh[lr * 136 +      c2] + bias[hcol];
      float gf = Gh[lr * 136 + 32 + c2] + bias[HID     + hcol];
      float gg = Gh[lr * 136 + 64 + c2] + bias[2 * HID + hcol];
      float go = Gh[lr * 136 + 96 + c2] + bias[3 * HID + hcol];
      float si = fsigmoid(gi);
      float sf = fsigmoid(gf);
      float tg = ftanh(gg);
      float so = fsigmoid(go);
      int   off = brow * HID + hcol;
      float cn = sf * cst[off] + si * tg;
      float hn = so * ftanh(cn);
      cst[off]     = cn;
      c_out[off]   = cn;
      h_out[off]   = hn;
      hout_bf[off] = f2bf(hn);
    }
  }
}

// ---- host driver ----------------------------------------------------------

extern "C" void kernel_launch(void* const* d_in, const int* in_sizes, int n_in,
                              void* d_out, int out_size, void* d_ws, size_t ws_size,
                              hipStream_t stream) {
  const int*   seq = (const int*)d_in[0];
  const float* emb = (const float*)d_in[1];
  const float* Wih = (const float*)d_in[2];
  const float* Whh = (const float*)d_in[3];
  const float* bih = (const float*)d_in[4];
  const float* bhh = (const float*)d_in[5];
  float* out = (float*)d_out;                   // h [64][512] then c [64][512]

  char* ws = (char*)d_ws;
  unsigned short* emb_bf = (unsigned short*)(ws);              // 16,777,216 B
  unsigned short* Wbf    = (unsigned short*)(ws + 16777216);   //  3,145,728 B
  float*          bias   = (float*)(ws + 19922944);            //      8,192 B
  unsigned short* hbf0   = (unsigned short*)(ws + 19931136);   //     65,536 B
  unsigned short* hbf1   = (unsigned short*)(ws + 19996672);   //     65,536 B
  float*          cst    = (float*)(ws + 20062208);            //    131,072 B

  prep_w<<<768, 256, 0, stream>>>(Wih, Whh, bih, bhh, Wbf, bias);
  prep_emb<<<4096, 256, 0, stream>>>(seq, emb, emb_bf);
  init_state<<<128, 256, 0, stream>>>((unsigned int*)hbf0, (unsigned int*)hbf1, cst);

  for (int s = 0; s < S_LEN; ++s) {
    const unsigned short* hi = (s & 1) ? hbf1 : hbf0;
    unsigned short*       ho = (s & 1) ? hbf0 : hbf1;
    lstm_step<<<16, 256, 0, stream>>>(
        emb_bf + (size_t)s * BATCH * EMB, hi, Wbf, bias, cst,
        ho, out, out + BATCH * HID);
  }
}
